// UNetPointGenerator_32865089749321
// MI455X (gfx1250) — matrix operand order repaired and checked by reading, not compiled, not run on hardware
//
#include <hip/hip_runtime.h>

#define HWDIM 256
#define NPIX  (HWDIM*HWDIM)
#define SENT  NPIX
#define NSEG  (NPIX+1)

typedef unsigned short u16;
typedef unsigned int   u32;

typedef __attribute__((ext_vector_type(16))) __bf16 v16bf;
typedef __attribute__((ext_vector_type(8)))  float  v8f;
typedef int vi4 __attribute__((vector_size(16)));   // matches async-LDS builtin param

#if __has_builtin(__builtin_amdgcn_global_load_async_to_lds_b128) && \
    __has_builtin(__builtin_amdgcn_s_wait_asynccnt)
#define HAS_ASYNC_LDS 1
#else
#define HAS_ASYNC_LDS 0
#endif

__device__ __forceinline__ float bf2f(u16 h){ u32 x=((u32)h)<<16; return __builtin_bit_cast(float,x); }
__device__ __forceinline__ u16 f2bf(float f){
  u32 x=__builtin_bit_cast(u32,f);
  x += 0x7fffu + ((x>>16)&1u);           // round-to-nearest-even
  return (u16)(x>>16);
}

union V16U { uint4 u[2]; v16bf v; };
__device__ __forceinline__ v16bf mkv16(uint4 lo, uint4 hi){ V16U t; t.u[0]=lo; t.u[1]=hi; return t.v; }

__device__ __forceinline__ v8f wmma_bf16(v16bf a, v16bf b, v8f c){
  return __builtin_amdgcn_wmma_f32_16x16x32_bf16(false, a, false, b, (short)0, c, false, false);
}

// ---------------------------------------------------------------------------
// One-time weight packing: OIHW f32 -> bf16 laid out in exact A-fragment order
//   wpk[band*taps*ncc*512 + (t*ncc + cc)*512 + lane*16 + e]
// taps = 9 (3x3 conv, t = ky*3+kx) or 4 (2x2 transposed conv, t = dy*2+dx).
// ---------------------------------------------------------------------------
__global__ void pack_w_kernel(const float* __restrict__ w, u16* __restrict__ wpk,
                              int Cin, int Cout, int taps) {
  int id = blockIdx.x * blockDim.x + threadIdx.x;
  int total = Cout * Cin * taps;
  if (id >= total) return;
  int ncc = Cin >> 5;
  int e = id & 15;
  int l = (id >> 4) & 31;
  int r = id >> 9;
  int cc = r % ncc;  r /= ncc;
  int t  = r % taps; int ob = r / taps;
  int o = ob * 16 + (l & 15);
  int offA = (l < 16) ? 0 : 16;                // A layout: contiguous 16 K
  int kk = offA + e;
  int c = cc * 32 + kk;
  wpk[id] = f2bf(w[((size_t)o * Cin + c) * taps + t]);
}

// ---------------------------------------------------------------------------
// Implicit-GEMM 3x3 SAME conv.
// Block: 256 threads = 8 waves. Macro-tile: 32 Cout x 256 pixels.
// Wave tile: 32 Cout (2 bands) x 32 pixels (2 tiles) -> 4 accumulators.
// A staged once per block into LDS, double-buffered via async global->LDS
// DMA (ASYNCcnt-tracked); B read per-lane from L2. Grids are exact.
// ---------------------------------------------------------------------------
__global__ __launch_bounds__(256) void conv3x3_wmma(
    const u16* __restrict__ act,      // [npix][Cin] bf16
    const u16* __restrict__ wpk,      // packed weights (taps=9)
    const float* __restrict__ bias,   // [Cout]
    u16* __restrict__ out,            // [npix][Cstride] bf16 (+coff)
    int H, int W, int Cin, int Cstride, int coff, int relu)
{
  const int tid   = threadIdx.x;
  const int lane  = tid & 31;
  const int wave  = tid >> 5;
  const int ptile = blockIdx.x * 256 + wave * 32;
  const int oband = blockIdx.y * 2;            // two 16-wide Cout bands
  const int ncc   = Cin >> 5;
  const int nchunk = 9 * ncc;

  const int n    = lane & 15;
  const int offB = (lane < 16) ? 0 : 16;       // B layout: contiguous 16 K

  int py[2], px[2];
  #pragma unroll
  for (int j = 0; j < 2; ++j) {
    int p = ptile + j * 16 + n;
    py[j] = p / W; px[j] = p % W;
  }

  v8f acc[2][2];
  #pragma unroll
  for (int i = 0; i < 2; ++i)
    #pragma unroll
    for (int j = 0; j < 2; ++j)
      acc[i][j] = (v8f){0.f,0.f,0.f,0.f,0.f,0.f,0.f,0.f};

#if HAS_ASYNC_LDS
  __shared__ u16 As[2][2][512];                // [buf][band][chunk frag data]
  // async stage of one chunk: 2 bands x 1KB = 128 lanes x b128
  auto stage = [&](int g, int buf) {
    if (tid < 128) {
      int i = tid >> 6, part = tid & 63;
      const u16* src = wpk + ((size_t)(oband + i) * nchunk + g) * 512 + part * 8;
      u16* dst = &As[buf][i][part * 8];
      __builtin_amdgcn_global_load_async_to_lds_b128(
          (vi4*)const_cast<u16*>(src), (vi4*)dst, 0, 0);
    }
  };
  stage(0, 0);
#else
  const u16* Ab[2];
  #pragma unroll
  for (int i = 0; i < 2; ++i)
    Ab[i] = wpk + (size_t)(oband + i) * nchunk * 512 + lane * 16;
#endif

  for (int t = 0; t < 9; ++t) {
    const int ty = t / 3, tx = t - ty * 3;
    const u16* bptr[2]; int bv[2];
    #pragma unroll
    for (int j = 0; j < 2; ++j) {
      int iy = py[j] + ty - 1, ix = px[j] + tx - 1;
      bv[j] = iy >= 0 && iy < H && ix >= 0 && ix < W;
      size_t basea = bv[j] ? ((size_t)(iy * W + ix) * Cin + offB) : 0;
      bptr[j] = act + basea;
    }
    for (int cc = 0; cc < ncc; ++cc) {
      const int g = t * ncc + cc;
      v16bf av[2];
#if HAS_ASYNC_LDS
      const int cur = g & 1;
      __builtin_amdgcn_s_wait_asynccnt(0);     // chunk g landed in LDS
      __syncthreads();                         // all waves done with other buf
      if (g + 1 < nchunk) stage(g + 1, cur ^ 1);  // overlap DMA with WMMAs
      #pragma unroll
      for (int i = 0; i < 2; ++i) {
        const uint4* ap = (const uint4*)(&As[cur][i][lane * 16]);
        av[i] = mkv16(ap[0], ap[1]);           // 2x ds_load_b128
      }
#else
      #pragma unroll
      for (int i = 0; i < 2; ++i) {
        uint4 a0 = *(const uint4*)(Ab[i]);
        uint4 a1 = *(const uint4*)(Ab[i] + 8);
        Ab[i] += 512;
        av[i] = mkv16(a0, a1);
      }
#endif
      uint4 b0[2], b1[2];
      #pragma unroll
      for (int j = 0; j < 2; ++j) {
        b0[j] = (uint4){0,0,0,0}; b1[j] = (uint4){0,0,0,0};
        if (bv[j]) {
          const uint4* bp = (const uint4*)(bptr[j] + (size_t)cc * 32);
          b0[j] = bp[0]; b1[j] = bp[1];
        }
      }
      #pragma unroll
      for (int i = 0; i < 2; ++i)
        #pragma unroll
        for (int j = 0; j < 2; ++j)
          acc[i][j] = wmma_bf16(av[i], mkv16(b0[j], b1[j]), acc[i][j]);
    }
  }

  // epilogue: C layout M = v + 8*(lane/16), N = lane%16 -> 8 contiguous chans
  #pragma unroll
  for (int i = 0; i < 2; ++i) {
    const int obase = (oband + i) * 16 + 8 * (lane >> 4);
    float bsum[8];
    #pragma unroll
    for (int v = 0; v < 8; ++v) bsum[v] = bias[obase + v];
    #pragma unroll
    for (int j = 0; j < 2; ++j) {
      int p = ptile + j * 16 + n;
      u16 res[8];
      #pragma unroll
      for (int v = 0; v < 8; ++v) {
        float y = acc[i][j][v] + bsum[v];
        if (relu) y = fmaxf(y, 0.f);
        res[v] = f2bf(y);
      }
      uint4 pk;
      pk.x = (u32)res[0] | ((u32)res[1] << 16);
      pk.y = (u32)res[2] | ((u32)res[3] << 16);
      pk.z = (u32)res[4] | ((u32)res[5] << 16);
      pk.w = (u32)res[6] | ((u32)res[7] << 16);
      *(uint4*)(&out[(size_t)p * Cstride + coff + obase]) = pk;
    }
  }
}

// ---------------------------------------------------------------------------
// Transposed conv k=2 s=2 as 4 parity-selected 1x1 GEMMs (blockIdx.z = tap).
// ---------------------------------------------------------------------------
__global__ __launch_bounds__(256) void upconv_wmma(
    const u16* __restrict__ in,       // [inW*inW][Cin] bf16
    const u16* __restrict__ wpk,      // packed (taps=4)
    const float* __restrict__ bias,
    u16* __restrict__ out,            // [4*inW*inW][Cstride] bf16
    int inW, int Cin, int Cstride)
{
  const int lane  = threadIdx.x & 31;
  const int wave  = threadIdx.x >> 5;
  const int qtile = blockIdx.x * 128 + wave * 16;
  const int otile = blockIdx.y * 16;
  const int t     = blockIdx.z;      // parity tap
  const int dy = t >> 1, dx = t & 1;
  const int ncc = Cin >> 5;

  const int n = lane & 15;
  const int q = qtile + n;
  const int y = q / inW, x = q % inW;
  const int offB = (lane < 16) ? 0 : 16;

  const u16* Ap = wpk + (size_t)(blockIdx.y * 4 + t) * ncc * 512 + lane * 16;
  const u16* bp = in + (size_t)q * Cin + offB;

  v8f acc = (v8f){0.f,0.f,0.f,0.f,0.f,0.f,0.f,0.f};
  for (int cc = 0; cc < ncc; ++cc) {
    uint4 a0 = *(const uint4*)Ap;
    uint4 a1 = *(const uint4*)(Ap + 8);
    Ap += 512;
    uint4 b0 = ((const uint4*)bp)[0];
    uint4 b1 = ((const uint4*)bp)[1];
    bp += 32;
    acc = wmma_bf16(mkv16(a0, a1), mkv16(b0, b1), acc);
  }

  const int obase = otile + 8 * (lane >> 4);
  const int op = (2*y + dy) * (2*inW) + (2*x + dx);
  u16 res[8];
  #pragma unroll
  for (int v = 0; v < 8; ++v) res[v] = f2bf(acc[v] + bias[obase + v]);
  uint4 pk;
  pk.x = (u32)res[0] | ((u32)res[1] << 16);
  pk.y = (u32)res[2] | ((u32)res[3] << 16);
  pk.z = (u32)res[4] | ((u32)res[5] << 16);
  pk.w = (u32)res[6] | ((u32)res[7] << 16);
  *(uint4*)(&out[(size_t)op * Cstride + obase]) = pk;
}

// ---------------------------------------------------------------------------
__global__ void resize_kernel(const float* __restrict__ img, u16* __restrict__ out) {
  int p = blockIdx.x * blockDim.x + threadIdx.x;
  if (p >= NPIX) return;
  int y = p >> 8, x = p & 255;
  int y0 = 4*y + 1, x0 = 4*x + 1;
  float v = 0.25f * (img[(size_t)y0*1024 + x0]     + img[(size_t)y0*1024 + x0+1] +
                     img[(size_t)(y0+1)*1024 + x0] + img[(size_t)(y0+1)*1024 + x0+1]);
  out[p] = f2bf(v * 255.0f);
}

__global__ void conv_c1_kernel(const u16* __restrict__ in, const float* __restrict__ w,
                               const float* __restrict__ b, u16* __restrict__ out) {
  int p = blockIdx.x * blockDim.x + threadIdx.x;
  if (p >= NPIX) return;
  int y = p >> 8, x = p & 255;
  float v[9];
  #pragma unroll
  for (int t = 0; t < 9; ++t) {
    int iy = y + t/3 - 1, ix = x + t%3 - 1;
    v[t] = (iy >= 0 && iy < HWDIM && ix >= 0 && ix < HWDIM) ? bf2f(in[iy*HWDIM + ix]) : 0.f;
  }
  for (int o = 0; o < 64; ++o) {
    float a = b[o];
    #pragma unroll
    for (int t = 0; t < 9; ++t) a += w[o*9 + t] * v[t];
    out[(size_t)p*64 + o] = f2bf(fmaxf(a, 0.f));
  }
}

__global__ void pool_kernel(const u16* __restrict__ in, u16* __restrict__ out,
                            int inW, int C, int total) {
  int id = blockIdx.x * blockDim.x + threadIdx.x;
  if (id >= total) return;
  int c = id % C, op = id / C;
  int ow = inW >> 1;
  int oy = op / ow, ox = op % ow;
  size_t b0 = ((size_t)(2*oy)*inW + 2*ox) * C + c;
  float m = bf2f(in[b0]);
  m = fmaxf(m, bf2f(in[b0 + C]));
  m = fmaxf(m, bf2f(in[b0 + (size_t)inW*C]));
  m = fmaxf(m, bf2f(in[b0 + (size_t)inW*C + C]));
  out[(size_t)op*C + c] = f2bf(m);
}

__global__ void copyskip_kernel(const u16* __restrict__ skip, u16* __restrict__ cat,
                                int C, int Ctot, int coff, int total) {
  int id = blockIdx.x * blockDim.x + threadIdx.x;
  if (id >= total) return;
  int c = id % C, p = id / C;
  cat[(size_t)p*Ctot + coff + c] = skip[(size_t)p*C + c];
}

__global__ void head_kernel(const u16* __restrict__ in, const float* __restrict__ w,
                            const float* __restrict__ b, float* __restrict__ prob,
                            float* __restrict__ dout) {
  int p = blockIdx.x * blockDim.x + threadIdx.x;
  if (p >= NPIX) return;
  float a = b[0];
  const u16* ip = &in[(size_t)p*64];
  #pragma unroll
  for (int c = 0; c < 64; ++c) a += bf2f(ip[c]) * w[c];
  float s = 1.0f / (1.0f + __expf(-a));
  prob[p] = s;
  dout[p] = s;
}

__global__ void rowmax_kernel(const float* __restrict__ in, float* __restrict__ out) {
  int p = blockIdx.x * blockDim.x + threadIdx.x;
  if (p >= NPIX) return;
  int y = p >> 8, x = p & 255;
  float m = -3.4e38f;
  for (int d = -5; d <= 5; ++d) { int xx = x + d; if (xx >= 0 && xx < HWDIM) m = fmaxf(m, in[y*HWDIM + xx]); }
  out[p] = m;
}
__global__ void colmax_kernel(const float* __restrict__ in, float* __restrict__ out) {
  int p = blockIdx.x * blockDim.x + threadIdx.x;
  if (p >= NPIX) return;
  int y = p >> 8, x = p & 255;
  float m = -3.4e38f;
  for (int d = -5; d <= 5; ++d) { int yy = y + d; if (yy >= 0 && yy < HWDIM) m = fmaxf(m, in[yy*HWDIM + x]); }
  out[p] = m;
}

__global__ void detect_kernel(const float* __restrict__ prob, const float* __restrict__ mf,
                              int* __restrict__ L) {
  int p = blockIdx.x * blockDim.x + threadIdx.x;
  if (p >= NPIX) return;
  L[p] = (prob[p] > 0.05f && mf[p] == prob[p]) ? p : SENT;
}

__global__ void prop_kernel(const int* __restrict__ Lin, int* __restrict__ Lout) {
  int p = blockIdx.x * blockDim.x + threadIdx.x;
  if (p >= NPIX) return;
  int l = Lin[p];
  if (l == SENT) { Lout[p] = SENT; return; }
  int y = p >> 8, x = p & 255;
  int m = l, t;
  if (y > 0)        { t = Lin[p - HWDIM]; if (t < m) m = t; }
  if (y < HWDIM-1)  { t = Lin[p + HWDIM]; if (t < m) m = t; }
  if (x > 0)        { t = Lin[p - 1];     if (t < m) m = t; }
  if (x < HWDIM-1)  { t = Lin[p + 1];     if (t < m) m = t; }
  Lout[p] = m;
}

__global__ void seginit_kernel(int* xmin, int* xmax, int* ymin, int* ymax, int* cnt) {
  int i = blockIdx.x * blockDim.x + threadIdx.x;
  if (i >= NSEG) return;
  xmin[i] = 0x7fffffff; ymin[i] = 0x7fffffff;
  xmax[i] = -1; ymax[i] = -1; cnt[i] = 0;
}

__global__ void segacc_kernel(const int* __restrict__ L, int* xmin, int* xmax,
                              int* ymin, int* ymax, int* cnt) {
  int p = blockIdx.x * blockDim.x + threadIdx.x;
  if (p >= NPIX) return;
  int id = L[p];
  if (id >= SENT) return;
  int y = p >> 8, x = p & 255;
  atomicMin(&xmin[id], x); atomicMax(&xmax[id], x);
  atomicMin(&ymin[id], y); atomicMax(&ymax[id], y);
  atomicAdd(&cnt[id], 1);
}

__global__ void finalize_kernel(const int* xmin, const int* xmax, const int* ymin,
                                const int* ymax, const int* cnt, float* __restrict__ out) {
  int i = blockIdx.x * blockDim.x + threadIdx.x;
  if (i >= NSEG) return;
  int v = (cnt[i] > 0) ? 1 : 0;
  float px = -1.f, py = -1.f;
  if (v) { px = (float)((xmin[i] + xmax[i]) >> 1); py = (float)((ymin[i] + ymax[i]) >> 1); }
  out[NPIX + 2*i + 0] = px;
  out[NPIX + 2*i + 1] = py;
  out[NPIX + 2*NSEG + i] = (float)v;
}

// ---------------------------------------------------------------------------
#define EW(n) dim3(((n) + 255) / 256), dim3(256)

extern "C" void kernel_launch(void* const* d_in, const int* in_sizes, int n_in,
                              void* d_out, int out_size, void* d_ws, size_t ws_size,
                              hipStream_t stream) {
  (void)in_sizes; (void)n_in; (void)out_size; (void)ws_size;
  const float* img = (const float*)d_in[0];
  auto F = [&](int i) { return (const float*)d_in[i]; };
  // sorted-pytree param order
  const float *bott_b1=F(1), *bott_b2=F(2), *bott_w1=F(3), *bott_w2=F(4);
  const float *d1_b1=F(5),  *d1_b2=F(6),  *d1_w1=F(7),  *d1_w2=F(8);
  const float *d2_b1=F(9),  *d2_b2=F(10), *d2_w1=F(11), *d2_w2=F(12);
  const float *d3_b1=F(13), *d3_b2=F(14), *d3_w1=F(15), *d3_w2=F(16);
  const float *d4_b1=F(17), *d4_b2=F(18), *d4_w1=F(19), *d4_w2=F(20);
  const float *e1_b1=F(21), *e1_b2=F(22), *e1_w1=F(23), *e1_w2=F(24);
  const float *e2_b1=F(25), *e2_b2=F(26), *e2_w1=F(27), *e2_w2=F(28);
  const float *e3_b1=F(29), *e3_b2=F(30), *e3_w1=F(31), *e3_w2=F(32);
  const float *e4_b1=F(33), *e4_b2=F(34), *e4_w1=F(35), *e4_w2=F(36);
  const float *out_b=F(37), *out_w=F(38);
  const float *u1_b=F(39), *u1_w=F(40), *u2_b=F(41), *u2_w=F(42);
  const float *u3_b=F(43), *u3_w=F(44), *u4_b=F(45), *u4_w=F(46);

  char* base = (char*)d_ws; size_t off = 0;
  auto alloc = [&](size_t bytes) -> void* {
    void* p = base + off; off = (off + bytes + 255) & ~(size_t)255; return p;
  };
  // activations (bf16 NHWC)
  u16* x256 = (u16*)alloc((size_t)NPIX * 2);
  u16* e1a = (u16*)alloc((size_t)NPIX*64*2);   u16* e1 = (u16*)alloc((size_t)NPIX*64*2);
  u16* p1  = (u16*)alloc((size_t)16384*64*2);
  u16* e2a = (u16*)alloc((size_t)16384*128*2); u16* e2 = (u16*)alloc((size_t)16384*128*2);
  u16* p2  = (u16*)alloc((size_t)4096*128*2);
  u16* e3a = (u16*)alloc((size_t)4096*256*2);  u16* e3 = (u16*)alloc((size_t)4096*256*2);
  u16* p3  = (u16*)alloc((size_t)1024*256*2);
  u16* e4a = (u16*)alloc((size_t)1024*512*2);  u16* e4 = (u16*)alloc((size_t)1024*512*2);
  u16* p4  = (u16*)alloc((size_t)256*512*2);
  u16* ba  = (u16*)alloc((size_t)256*1024*2);  u16* bb = (u16*)alloc((size_t)256*1024*2);
  u16* cat4= (u16*)alloc((size_t)1024*1024*2);
  u16* d4a = (u16*)alloc((size_t)1024*512*2);  u16* d4 = (u16*)alloc((size_t)1024*512*2);
  u16* cat3= (u16*)alloc((size_t)4096*512*2);
  u16* d3a = (u16*)alloc((size_t)4096*256*2);  u16* d3 = (u16*)alloc((size_t)4096*256*2);
  u16* cat2= (u16*)alloc((size_t)16384*256*2);
  u16* d2a = (u16*)alloc((size_t)16384*128*2); u16* d2 = (u16*)alloc((size_t)16384*128*2);
  u16* cat1= (u16*)alloc((size_t)NPIX*128*2);
  u16* d1a = (u16*)alloc((size_t)NPIX*64*2);   u16* d1 = (u16*)alloc((size_t)NPIX*64*2);
  float* prob = (float*)alloc((size_t)NPIX*4);
  float* tmpf = (float*)alloc((size_t)NPIX*4);
  float* mf   = (float*)alloc((size_t)NPIX*4);
  int* L0 = (int*)alloc((size_t)NPIX*4);
  int* L1 = (int*)alloc((size_t)NPIX*4);
  int* sxmin = (int*)alloc((size_t)NSEG*4); int* sxmax = (int*)alloc((size_t)NSEG*4);
  int* symin = (int*)alloc((size_t)NSEG*4); int* symax = (int*)alloc((size_t)NSEG*4);
  int* scnt  = (int*)alloc((size_t)NSEG*4);
  float* outf = (float*)d_out;

  // weight packing (once per call; L2-resident thereafter)
  auto packw = [&](const float* w, int Cin, int Cout, int taps) -> u16* {
    size_t n = (size_t)Cout * Cin * taps;
    u16* p = (u16*)alloc(n * 2);
    pack_w_kernel<<<EW((int)n), 0, stream>>>(w, p, Cin, Cout, taps);
    return p;
  };
  u16* pk_e1c2 = packw(e1_w2, 64, 64, 9);
  u16* pk_e2c1 = packw(e2_w1, 64, 128, 9);   u16* pk_e2c2 = packw(e2_w2, 128, 128, 9);
  u16* pk_e3c1 = packw(e3_w1, 128, 256, 9);  u16* pk_e3c2 = packw(e3_w2, 256, 256, 9);
  u16* pk_e4c1 = packw(e4_w1, 256, 512, 9);  u16* pk_e4c2 = packw(e4_w2, 512, 512, 9);
  u16* pk_bc1  = packw(bott_w1, 512, 1024, 9); u16* pk_bc2 = packw(bott_w2, 1024, 1024, 9);
  u16* pk_d4c1 = packw(d4_w1, 1024, 512, 9); u16* pk_d4c2 = packw(d4_w2, 512, 512, 9);
  u16* pk_d3c1 = packw(d3_w1, 512, 256, 9);  u16* pk_d3c2 = packw(d3_w2, 256, 256, 9);
  u16* pk_d2c1 = packw(d2_w1, 256, 128, 9);  u16* pk_d2c2 = packw(d2_w2, 128, 128, 9);
  u16* pk_d1c1 = packw(d1_w1, 128, 64, 9);   u16* pk_d1c2 = packw(d1_w2, 64, 64, 9);
  u16* pk_u4 = packw(u4_w, 1024, 512, 4);
  u16* pk_u3 = packw(u3_w, 512, 256, 4);
  u16* pk_u2 = packw(u2_w, 256, 128, 4);
  u16* pk_u1 = packw(u1_w, 128, 64, 4);

  auto conv = [&](const u16* act, const u16* wpk, const float* b, u16* out_,
                  int Hh, int Ww, int Cin, int Cout, int Cstride, int coff) {
    dim3 g((Hh * Ww) / 256, Cout / 32);
    conv3x3_wmma<<<g, 256, 0, stream>>>(act, wpk, b, out_, Hh, Ww, Cin, Cstride, coff, 1);
  };
  auto upc = [&](const u16* in, const u16* wpk, const float* b, u16* out_,
                 int inW, int Cin, int Cout, int Cstride) {
    dim3 g((inW * inW) / 128, Cout / 16, 4);
    upconv_wmma<<<g, 256, 0, stream>>>(in, wpk, b, out_, inW, Cin, Cstride);
  };

  // ---- encoder ----
  resize_kernel<<<EW(NPIX), 0, stream>>>(img, x256);
  conv_c1_kernel<<<EW(NPIX), 0, stream>>>(x256, e1_w1, e1_b1, e1a);
  conv(e1a, pk_e1c2, e1_b2, e1, 256, 256, 64, 64, 64, 0);
  pool_kernel<<<EW(16384*64), 0, stream>>>(e1, p1, 256, 64, 16384*64);
  conv(p1,  pk_e2c1, e2_b1, e2a, 128, 128, 64, 128, 128, 0);
  conv(e2a, pk_e2c2, e2_b2, e2,  128, 128, 128, 128, 128, 0);
  pool_kernel<<<EW(4096*128), 0, stream>>>(e2, p2, 128, 128, 4096*128);
  conv(p2,  pk_e3c1, e3_b1, e3a, 64, 64, 128, 256, 256, 0);
  conv(e3a, pk_e3c2, e3_b2, e3,  64, 64, 256, 256, 256, 0);
  pool_kernel<<<EW(1024*256), 0, stream>>>(e3, p3, 64, 256, 1024*256);
  conv(p3,  pk_e4c1, e4_b1, e4a, 32, 32, 256, 512, 512, 0);
  conv(e4a, pk_e4c2, e4_b2, e4,  32, 32, 512, 512, 512, 0);
  pool_kernel<<<EW(256*512), 0, stream>>>(e4, p4, 32, 512, 256*512);
  conv(p4, pk_bc1, bott_b1, ba, 16, 16, 512, 1024, 1024, 0);
  conv(ba, pk_bc2, bott_b2, bb, 16, 16, 1024, 1024, 1024, 0);

  // ---- decoder ----
  upc(bb, pk_u4, u4_b, cat4, 16, 1024, 512, 1024);
  copyskip_kernel<<<EW(1024*512), 0, stream>>>(e4, cat4, 512, 1024, 512, 1024*512);
  conv(cat4, pk_d4c1, d4_b1, d4a, 32, 32, 1024, 512, 512, 0);
  conv(d4a,  pk_d4c2, d4_b2, d4,  32, 32, 512, 512, 512, 0);

  upc(d4, pk_u3, u3_b, cat3, 32, 512, 256, 512);
  copyskip_kernel<<<EW(4096*256), 0, stream>>>(e3, cat3, 256, 512, 256, 4096*256);
  conv(cat3, pk_d3c1, d3_b1, d3a, 64, 64, 512, 256, 256, 0);
  conv(d3a,  pk_d3c2, d3_b2, d3,  64, 64, 256, 256, 256, 0);

  upc(d3, pk_u2, u2_b, cat2, 64, 256, 128, 256);
  copyskip_kernel<<<EW(16384*128), 0, stream>>>(e2, cat2, 128, 256, 128, 16384*128);
  conv(cat2, pk_d2c1, d2_b1, d2a, 128, 128, 256, 128, 128, 0);
  conv(d2a,  pk_d2c2, d2_b2, d2,  128, 128, 128, 128, 128, 0);

  upc(d2, pk_u1, u1_b, cat1, 128, 128, 64, 128);
  copyskip_kernel<<<EW(NPIX*64), 0, stream>>>(e1, cat1, 64, 128, 64, NPIX*64);
  conv(cat1, pk_d1c1, d1_b1, d1a, 256, 256, 128, 64, 64, 0);
  conv(d1a,  pk_d1c2, d1_b2, d1,  256, 256, 64, 64, 64, 0);

  // ---- head + peak detection + connected components ----
  head_kernel<<<EW(NPIX), 0, stream>>>(d1, out_w, out_b, prob, outf);
  rowmax_kernel<<<EW(NPIX), 0, stream>>>(prob, tmpf);
  colmax_kernel<<<EW(NPIX), 0, stream>>>(tmpf, mf);
  detect_kernel<<<EW(NPIX), 0, stream>>>(prob, mf, L0);
  int* A = L0; int* B = L1;
  for (int it = 0; it < 64; ++it) {
    prop_kernel<<<EW(NPIX), 0, stream>>>(A, B);
    int* t = A; A = B; B = t;
  }
  seginit_kernel<<<EW(NSEG), 0, stream>>>(sxmin, sxmax, symin, symax, scnt);
  segacc_kernel<<<EW(NPIX), 0, stream>>>(A, sxmin, sxmax, symin, symax, scnt);
  finalize_kernel<<<EW(NSEG), 0, stream>>>(sxmin, sxmax, symin, symax, scnt, outf);
}
